// CausalMultiHeadAttention_55138790146348
// MI455X (gfx1250) — matrix operand order repaired and checked by reading, not compiled, not run on hardware
//
#include <hip/hip_runtime.h>
#include <hip/hip_bf16.h>

typedef _Float16 f16;
typedef __attribute__((ext_vector_type(16))) _Float16 v16h;
typedef __attribute__((ext_vector_type(8)))  _Float16 v8h;
typedef __attribute__((ext_vector_type(8)))  float    v8f;

constexpr int BATCH = 4;
constexpr int T     = 2048;
constexpr int D     = 1024;
constexpr int H     = 16;
constexpr int DH    = 64;         // head dim
constexpr int MROW  = BATCH * T;  // 8192 rows for projection GEMMs

__device__ __forceinline__ v8f vzero8() {
  v8f z = {0.f,0.f,0.f,0.f,0.f,0.f,0.f,0.f};
  return z;
}

__device__ __forceinline__ v8f wmma16(v16h a, v16h b, v8f c) {
  // D = A(16x32 f16) * B(32x16 f16) + C(16x16 f32)
  return __builtin_amdgcn_wmma_f32_16x16x32_f16(
      /*neg_a=*/false, a, /*neg_b=*/false, b,
      /*c_mod=*/(short)0, c, /*reuse_a=*/false, /*reuse_b=*/false);
}

// A-matrix fragment (16x32, MxK) from row-major tile with leading dim ld.
// Lane l: m = l%16, half = l/16. VGPRs 0..3 -> K = half*8..+7, 4..7 -> K = 16+half*8..+7
__device__ __forceinline__ v16h load_afrag(const f16* tile, int ld) {
  const int lane = threadIdx.x & 31;
  const int m    = lane & 15;
  const int half = lane >> 4;
  const f16* p = tile + (size_t)m * ld + half * 8;
  v8h lo = *reinterpret_cast<const v8h*>(p);
  v8h hi = *reinterpret_cast<const v8h*>(p + 16);
  v16h r;
#pragma unroll
  for (int i = 0; i < 8; ++i) { r[i] = lo[i]; r[8 + i] = hi[i]; }
  return r;
}

// B-matrix fragment (32x16, KxN) from B-transposed tile Bt[n][k] row-major, ld.
// Lane l: n = l%16, half = l/16. VGPRs 0..3 -> K = half*8..+7, 4..7 -> K = 16+half*8..+7
__device__ __forceinline__ v16h load_bfrag(const f16* tile, int ld) {
  const int lane = threadIdx.x & 31;
  const int n    = lane & 15;
  const int half = lane >> 4;
  const f16* p = tile + (size_t)n * ld + half * 8;
  v8h lo = *reinterpret_cast<const v8h*>(p);
  v8h hi = *reinterpret_cast<const v8h*>(p + 16);
  v16h r;
#pragma unroll
  for (int i = 0; i < 8; ++i) { r[i] = lo[i]; r[8 + i] = hi[i]; }
  return r;
}

// ---------------------------------------------------------------------------
// fp32 -> f16 conversion
// ---------------------------------------------------------------------------
__global__ void cvt_f32_to_f16(const float* __restrict__ in,
                               f16* __restrict__ out, int n) {
  int i = blockIdx.x * blockDim.x + threadIdx.x;
  if (i < n) out[i] = (f16)in[i];
}

// ---------------------------------------------------------------------------
// GEMM: out[m][n] = sum_k A[m][k] * Bt[n][k]    (nn.Linear: A @ Bt^T)
// One wave per block; wave computes a 32(M) x 64(N) tile, software-pipelined
// one k-step (32) ahead. __launch_bounds__(32,1): single-wave workgroup with
// occupancy floor 1 so the ~200 live VGPRs stay in registers (no scratch
// spills in the hot loop; wave32 allows up to 1024 VGPRs/wave).
// outMode: 0 = fp32 row-major, 1 = f16 row-major,
//          2 = f16 per-head transposed VT[b][h][dh][t]  (for attention V)
// ---------------------------------------------------------------------------
__global__ void __launch_bounds__(32, 1)
gemm_xWt(const f16* __restrict__ A, const f16* __restrict__ Bt,
         float* __restrict__ outF, f16* __restrict__ outB,
         int K, int lda, int ldb, int ldo, int outMode) {
  const int n0 = blockIdx.x * 64;
  const int m0 = blockIdx.y * 32;
  const int lane = threadIdx.x & 31;
  const int n    = lane & 15;
  const int half = lane >> 4;

  v8f acc[2][4];
#pragma unroll
  for (int i = 0; i < 2; ++i)
#pragma unroll
    for (int j = 0; j < 4; ++j) acc[i][j] = vzero8();

  const f16* Arow0 = A + (size_t)m0 * lda;
  const f16* Arow1 = A + (size_t)(m0 + 16) * lda;
  const f16* Brow0 = Bt + (size_t)(n0 +  0) * ldb;
  const f16* Brow1 = Bt + (size_t)(n0 + 16) * ldb;
  const f16* Brow2 = Bt + (size_t)(n0 + 32) * ldb;
  const f16* Brow3 = Bt + (size_t)(n0 + 48) * ldb;

  // prologue: fragments for k0 = 0
  v16h a0 = load_afrag(Arow0, lda);
  v16h a1 = load_afrag(Arow1, lda);
  v16h b0 = load_bfrag(Brow0, ldb);
  v16h b1 = load_bfrag(Brow1, ldb);
  v16h b2 = load_bfrag(Brow2, ldb);
  v16h b3 = load_bfrag(Brow3, ldb);

  for (int k0 = 0; k0 < K - 32; k0 += 32) {
    const int kn = k0 + 32;
    // issue next-step loads first; they retire while WMMAs run
    v16h a0n = load_afrag(Arow0 + kn, lda);
    v16h a1n = load_afrag(Arow1 + kn, lda);
    v16h b0n = load_bfrag(Brow0 + kn, ldb);
    v16h b1n = load_bfrag(Brow1 + kn, ldb);
    v16h b2n = load_bfrag(Brow2 + kn, ldb);
    v16h b3n = load_bfrag(Brow3 + kn, ldb);

    acc[0][0] = wmma16(a0, b0, acc[0][0]);
    acc[1][0] = wmma16(a1, b0, acc[1][0]);
    acc[0][1] = wmma16(a0, b1, acc[0][1]);
    acc[1][1] = wmma16(a1, b1, acc[1][1]);
    acc[0][2] = wmma16(a0, b2, acc[0][2]);
    acc[1][2] = wmma16(a1, b2, acc[1][2]);
    acc[0][3] = wmma16(a0, b3, acc[0][3]);
    acc[1][3] = wmma16(a1, b3, acc[1][3]);

    a0 = a0n; a1 = a1n; b0 = b0n; b1 = b1n; b2 = b2n; b3 = b3n;
  }
  // epilogue k-step
  acc[0][0] = wmma16(a0, b0, acc[0][0]);
  acc[1][0] = wmma16(a1, b0, acc[1][0]);
  acc[0][1] = wmma16(a0, b1, acc[0][1]);
  acc[1][1] = wmma16(a1, b1, acc[1][1]);
  acc[0][2] = wmma16(a0, b2, acc[0][2]);
  acc[1][2] = wmma16(a1, b2, acc[1][2]);
  acc[0][3] = wmma16(a0, b3, acc[0][3]);
  acc[1][3] = wmma16(a1, b3, acc[1][3]);

#pragma unroll
  for (int mi = 0; mi < 2; ++mi) {
#pragma unroll
    for (int j = 0; j < 4; ++j) {
#pragma unroll
      for (int r = 0; r < 8; ++r) {
        int row = m0 + mi * 16 + half * 8 + r;   // C layout: M = half*8 + r
        int col = n0 + j * 16 + n;               // N = lane%16
        float v = acc[mi][j][r];
        if (outMode == 0) {
          outF[(size_t)row * ldo + col] = v;
        } else if (outMode == 1) {
          outB[(size_t)row * ldo + col] = (f16)v;
        } else {
          // V projection stored pre-transposed per head: VT[b][h][dh][t]
          int bb = row / T, t  = row % T;
          int hh = col / DH, dh = col % DH;
          outB[(((size_t)bb * H + hh) * DH + dh) * T + t] = (f16)v;
        }
      }
    }
  }
}

// ---------------------------------------------------------------------------
// Fused causal flash-attention. One wave per block; wave owns 16 query rows
// of one (batch, head). Online softmax over key blocks of 32.
// Q,K are f16 [B][T][D] (head h at column h*DH); VT is f16 [B][H][DH][T].
// Only LDS use: 1KB probs tile for the C-layout -> A-layout conversion.
// ---------------------------------------------------------------------------
__global__ void __launch_bounds__(32, 1)
attn_flash(const f16* __restrict__ Q, const f16* __restrict__ K,
           const f16* __restrict__ VT, f16* __restrict__ O) {
  __shared__ f16 Ps[16 * 32];   // probs tile 16(q) x 32(key), row-major

  const int qblk = blockIdx.x;
  const int h    = blockIdx.y;
  const int b    = blockIdx.z;
  const int lane = threadIdx.x & 31;
  const int n    = lane & 15;
  const int half = lane >> 4;
  const int qbase = qblk * 16;
  const float scale = 0.125f;   // 1/sqrt(64)

  const f16* Qbase = Q + ((size_t)b * T + qbase) * D + h * DH;
  const f16* Vhead = VT + ((size_t)b * H + h) * DH * T;
  v16h q0 = load_afrag(Qbase, D);        // dh 0..31
  v16h q1 = load_afrag(Qbase + 32, D);   // dh 32..63

  v8f acc[4];
#pragma unroll
  for (int j = 0; j < 4; ++j) acc[j] = vzero8();
  float mrow[8], lrow[8];
#pragma unroll
  for (int r = 0; r < 8; ++r) { mrow[r] = -1e30f; lrow[r] = 0.f; }

  for (int kb = 0; kb < qbase + 16; kb += 32) {
    // ---- batch all K-fragment loads, then S = Q @ K^T (two 16-key tiles) ----
    const f16* Kb0 = K + ((size_t)b * T + kb) * D + h * DH;
    const f16* Kb1 = K + ((size_t)b * T + kb + 16) * D + h * DH;
    v16h kf00 = load_bfrag(Kb0,      D);
    v16h kf01 = load_bfrag(Kb0 + 32, D);
    v16h kf10 = load_bfrag(Kb1,      D);
    v16h kf11 = load_bfrag(Kb1 + 32, D);
    v8f s0 = vzero8(), s1 = vzero8();
    s0 = wmma16(q0, kf00, s0);
    s0 = wmma16(q1, kf01, s0);
    s1 = wmma16(q0, kf10, s1);
    s1 = wmma16(q1, kf11, s1);

    // ---- issue V-fragment loads now; softmax VALU work hides their latency ----
    v16h vf0 = load_bfrag(Vhead + (size_t)( 0) * T + kb, T);
    v16h vf1 = load_bfrag(Vhead + (size_t)(16) * T + kb, T);
    v16h vf2 = load_bfrag(Vhead + (size_t)(32) * T + kb, T);
    v16h vf3 = load_bfrag(Vhead + (size_t)(48) * T + kb, T);

    // ---- causal mask + scale + online softmax update ----
#pragma unroll
    for (int r = 0; r < 8; ++r) {
      const int row  = qbase + half * 8 + r;
      const int key0 = kb + n;
      const int key1 = key0 + 16;
      float a = (key0 <= row) ? s0[r] * scale : -1e30f;
      float c = (key1 <= row) ? s1[r] * scale : -1e30f;
      s0[r] = a; s1[r] = c;
      float mx = fmaxf(a, c);
#pragma unroll
      for (int off = 1; off < 16; off <<= 1)
        mx = fmaxf(mx, __shfl_xor(mx, off, 32));   // stays in 16-lane row group
      float mn = fmaxf(mrow[r], mx);

      float p0 = __expf(s0[r] - mn);
      float p1 = __expf(s1[r] - mn);
      float corr = __expf(mrow[r] - mn);
      float rs = p0 + p1;
#pragma unroll
      for (int off = 1; off < 16; off <<= 1)
        rs += __shfl_xor(rs, off, 32);
      lrow[r] = lrow[r] * corr + rs;
      mrow[r] = mn;
#pragma unroll
      for (int j = 0; j < 4; ++j) acc[j][r] *= corr;

      const int m = half * 8 + r;                  // C-layout row inside tile
      Ps[m * 32 + n]      = (f16)p0;
      Ps[m * 32 + 16 + n] = (f16)p1;
    }

    asm volatile("s_wait_dscnt 0" ::: "memory");   // Ps stores visible (same wave)

    // ---- acc += P(16x32) @ V(32x64) ----
    v16h pa = load_afrag(Ps, 32);
    acc[0] = wmma16(pa, vf0, acc[0]);
    acc[1] = wmma16(pa, vf1, acc[1]);
    acc[2] = wmma16(pa, vf2, acc[2]);
    acc[3] = wmma16(pa, vf3, acc[3]);
  }

  // ---- normalize and write O (row-major [B][T][D]) ----
#pragma unroll
  for (int r = 0; r < 8; ++r) {
    const float inv = 1.0f / lrow[r];
    const int row = qbase + half * 8 + r;
    f16* op = O + ((size_t)b * T + row) * D + h * DH;
#pragma unroll
    for (int j = 0; j < 4; ++j) op[j * 16 + n] = (f16)(acc[j][r] * inv);
  }
}

// ---------------------------------------------------------------------------
// Host launcher
// ---------------------------------------------------------------------------
extern "C" void kernel_launch(void* const* d_in, const int* in_sizes, int n_in,
                              void* d_out, int out_size, void* d_ws, size_t ws_size,
                              hipStream_t stream) {
  const float* x  = (const float*)d_in[0];   // [4,2048,1024]
  const float* Wq = (const float*)d_in[1];   // [1024,1024]
  const float* Wk = (const float*)d_in[2];
  const float* Wv = (const float*)d_in[3];
  const float* Wo = (const float*)d_in[4];

  const size_t NX = (size_t)MROW * D;        // 8,388,608
  const size_t NW = (size_t)D * D;           // 1,048,576

  f16* ws = (f16*)d_ws;
  size_t off = 0;
  f16* xb  = ws + off; off += NX;
  f16* Wqb = ws + off; off += NW;
  f16* Wkb = ws + off; off += NW;
  f16* Wvb = ws + off; off += NW;
  f16* Wob = ws + off; off += NW;
  f16* Qb  = ws + off; off += NX;
  f16* Kb  = ws + off; off += NX;
  f16* VTb = ws + off; off += NX;   // [B][H][DH][T]
  f16* Ob  = ws + off; off += NX;   // total ~92 MB f16 scratch

  // fp32 -> f16
  cvt_f32_to_f16<<<(int)((NX + 255) / 256), 256, 0, stream>>>(x,  xb,  (int)NX);
  cvt_f32_to_f16<<<(int)((NW + 255) / 256), 256, 0, stream>>>(Wq, Wqb, (int)NW);
  cvt_f32_to_f16<<<(int)((NW + 255) / 256), 256, 0, stream>>>(Wk, Wkb, (int)NW);
  cvt_f32_to_f16<<<(int)((NW + 255) / 256), 256, 0, stream>>>(Wv, Wvb, (int)NW);
  cvt_f32_to_f16<<<(int)((NW + 255) / 256), 256, 0, stream>>>(Wo, Wob, (int)NW);

  // projections: [8192,1024] @ W^T ; V written pre-transposed per head
  dim3 ggrid(D / 64, MROW / 32);   // (16, 256)
  gemm_xWt<<<ggrid, 32, 0, stream>>>(xb, Wqb, nullptr, Qb,  D, D, D, D, 1);
  gemm_xWt<<<ggrid, 32, 0, stream>>>(xb, Wkb, nullptr, Kb,  D, D, D, D, 1);
  gemm_xWt<<<ggrid, 32, 0, stream>>>(xb, Wvb, nullptr, VTb, D, D, D, D, 2);

  // fused causal attention per (q-block, head, batch)
  attn_flash<<<dim3(T / 16, H, BATCH), 32, 0, stream>>>(Qb, Kb, VTb, Ob);

  // output projection -> fp32 d_out
  gemm_xWt<<<ggrid, 32, 0, stream>>>(Ob, Wob, (float*)d_out, nullptr,
                                     D, D, D, D, 0);
}
